// DiffAttn_85040352461213
// MI455X (gfx1250) — matrix operand order repaired and checked by reading, not compiled, not run on hardware
//
#include <hip/hip_runtime.h>
#include <hip/hip_bf16.h>

typedef __attribute__((ext_vector_type(16))) __bf16 v16bf;
typedef __attribute__((ext_vector_type(8)))  float  v8f;

#define HD 128
#define DMODEL 2048
#define TLEN 4096
#define BN 32
#define BM 64
#define LAMBDA_INIT 0.783605766532f
#define LOG2E 1.44269504088896f
#define QSCALE (0.125f * LOG2E)      // 1/sqrt(64) folded with log2(e): scores in log2 domain

#if defined(__gfx1250__) && __has_builtin(__builtin_amdgcn_global_load_async_to_lds_b128) && __has_builtin(__builtin_amdgcn_s_wait_asynccnt)
#define HAVE_ASYNC 1
typedef int v4i __attribute__((vector_size(4 * sizeof(int))));
typedef __attribute__((address_space(1))) v4i gv4i;   // global-memory side
typedef __attribute__((address_space(3))) v4i lv4i;   // LDS side
#else
#define HAVE_ASYNC 0
#endif

static __device__ __forceinline__ __bf16 f2bf(float f) {
    return static_cast<__bf16>(f);   // native v_cvt_pk_bf16_f32 on gfx1250
}

// ---------------------------------------------------------------------------
// Kernel 1: projection  q/k/v = x @ w   (f32 in -> bf16 out, q pre-scaled by
// SCALE*log2(e) so attention runs a pure exp2 softmax). Double-buffered w
// staging: next 32-deep chunk is register-staged while WMMAs consume current.
// grid: (8192/64, 1, 3), block 128 (4 waves). Wave w: rows 16w..16w+15.
// ---------------------------------------------------------------------------
__global__ void proj_kernel(const float* __restrict__ x,
                            const float* __restrict__ wq,
                            const float* __restrict__ wk,
                            const float* __restrict__ wv,
                            __bf16* __restrict__ qo,
                            __bf16* __restrict__ ko,
                            __bf16* __restrict__ vo) {
    const float* w = (blockIdx.z == 0) ? wq : (blockIdx.z == 1) ? wk : wv;
    __bf16* outp   = (blockIdx.z == 0) ? qo : (blockIdx.z == 1) ? ko : vo;
    const float scale = (blockIdx.z == 0) ? QSCALE : 1.0f;

    __shared__ __bf16 wT[2][HD][BN + 8];   // double-buffered transposed w chunk

    const int tid  = threadIdx.x;
    const int wave = tid >> 5;
    const int lane = tid & 31;
    const int half = lane >> 4;
    const int l16  = lane & 15;
    const int rowBase = blockIdx.x * 64 + wave * 16;

    auto loadW = [&](int kc, float4* wr) {
        #pragma unroll
        for (int c = 0; c < 8; ++c) {
            int idx = tid + c * 128;            // 1024 float4s per 32x128 chunk
            int kk  = idx >> 5;                 // 32 float4 per row
            int n4  = (idx & 31) * 4;
            wr[c] = *(const float4*)&w[(size_t)(kc * BN + kk) * HD + n4];
        }
    };
    auto storeW = [&](const float4* wr, int buf) {
        #pragma unroll
        for (int c = 0; c < 8; ++c) {
            int idx = tid + c * 128;
            int kk  = idx >> 5;
            int n4  = (idx & 31) * 4;
            wT[buf][n4 + 0][kk] = f2bf(wr[c].x);
            wT[buf][n4 + 1][kk] = f2bf(wr[c].y);
            wT[buf][n4 + 2][kk] = f2bf(wr[c].z);
            wT[buf][n4 + 3][kk] = f2bf(wr[c].w);
        }
    };

    v8f acc[8];
    #pragma unroll
    for (int i = 0; i < 8; ++i)
        #pragma unroll
        for (int j = 0; j < 8; ++j) acc[i][j] = 0.0f;

    float4 wr[8], wr2[8];
    loadW(0, wr);

    for (int kc = 0; kc < DMODEL / BN; ++kc) {
        const int c   = kc * BN;
        const int buf = kc & 1;
        storeW(wr, buf);
        if (kc + 1 < DMODEL / BN) loadW(kc + 1, wr2);
        __syncthreads();

        // A fragment of x (16 rows x 32 deep), converted on the fly
        v16bf a;
        {
            const float* xr = x + (size_t)(rowBase + l16) * DMODEL + c + half * 16;
            #pragma unroll
            for (int j = 0; j < 8; ++j) { a[j] = f2bf(xr[j]); a[8 + j] = f2bf(xr[8 + j]); }
        }
        #pragma unroll
        for (int n0 = 0; n0 < 8; ++n0) {
            v16bf b;
            const __bf16* wp = &wT[buf][n0 * 16 + l16][half * 16];
            #pragma unroll
            for (int j = 0; j < 16; ++j) b[j] = wp[j];
            acc[n0] = __builtin_amdgcn_wmma_f32_16x16x32_bf16(
                false, a, false, b, (short)0, acc[n0], false, false);
        }
        __syncthreads();
        #pragma unroll
        for (int i = 0; i < 8; ++i) wr[i] = wr2[i];
    }

    // write-out: element (M = r + 8*half, N = n0*16 + l16)
    #pragma unroll
    for (int n0 = 0; n0 < 8; ++n0)
        #pragma unroll
        for (int r = 0; r < 8; ++r) {
            int m = rowBase + r + 8 * half;
            outp[(size_t)m * HD + n0 * 16 + l16] = f2bf(acc[n0][r] * scale);
        }
}

// ---------------------------------------------------------------------------
// Kernel 2: lambda scalar
// ---------------------------------------------------------------------------
__global__ void lambda_kernel(const float* __restrict__ lq1, const float* __restrict__ lq2,
                              const float* __restrict__ lk1, const float* __restrict__ lk2,
                              float* __restrict__ lam) {
    __shared__ float s1[64], s2[64];
    int t = threadIdx.x;
    s1[t] = lq1[t] * lk1[t];
    s2[t] = lq2[t] * lk2[t];
    __syncthreads();
    for (int o = 32; o > 0; o >>= 1) {
        if (t < o) { s1[t] += s1[t + o]; s2[t] += s2[t + o]; }
        __syncthreads();
    }
    if (t == 0) *lam = __expf(s1[0]) - __expf(s2[0]) + LAMBDA_INIT;
}

// ---------------------------------------------------------------------------
// Kernel 3: causal differential flash attention, double-buffered pipeline:
// K tiles via async LDS-DMA (ASYNCcnt), V tiles register-staged + transposed.
// grid: (4096/64, 2), block 128 (4 waves, 16 query rows each)
// ---------------------------------------------------------------------------
__global__ void diffattn_kernel(const __bf16* __restrict__ Q,
                                const __bf16* __restrict__ K,
                                const __bf16* __restrict__ V,
                                const float* __restrict__ lam_p,
                                const float* __restrict__ ln_w,
                                float* __restrict__ out) {
    __shared__ __bf16 Ks[2][BN][HD + 16];   // 32 keys x 128 dims, double-buffered
    __shared__ __bf16 VT[2][HD][BN + 8];    // V transposed: [dim][key]
    __shared__ __bf16 Ps[4][16][BN + 8];    // per-wave P bounce buffer

    const int tid  = threadIdx.x;
    const int wave = tid >> 5;
    const int lane = tid & 31;
    const int half = lane >> 4;
    const int l16  = lane & 15;
    const int batchBase = blockIdx.y * TLEN;
    const int qb = blockIdx.x * BM;
    const int qw = qb + wave * 16;          // this wave's query rows
    const float lam = *lam_p;

    auto issueK = [&](int t, int buf) {     // async global->LDS copy of K tile t
#if HAVE_ASYNC
        #pragma unroll
        for (int c = 0; c < 4; ++c) {
            int i  = tid + c * 128;
            int kk = i >> 4;
            int n8 = (i & 15) * 8;
            size_t g = (size_t)(batchBase + t * BN + kk) * HD + n8;
            __builtin_amdgcn_global_load_async_to_lds_b128(
                (gv4i*)(K + g), (lv4i*)&Ks[buf][kk][n8], 0, 0);
        }
#else
        #pragma unroll
        for (int c = 0; c < 4; ++c) {
            int i  = tid + c * 128;
            int kk = i >> 4;
            int n8 = (i & 15) * 8;
            size_t g = (size_t)(batchBase + t * BN + kk) * HD + n8;
            *(uint4*)&Ks[buf][kk][n8] = *(const uint4*)(K + g);
        }
#endif
    };
    auto loadV = [&](int t, uint4* vr) {
        #pragma unroll
        for (int c = 0; c < 4; ++c) {
            int i  = tid + c * 128;
            int kk = i >> 4;
            int n8 = (i & 15) * 8;
            vr[c] = *(const uint4*)(V + (size_t)(batchBase + t * BN + kk) * HD + n8);
        }
    };
    auto storeV = [&](const uint4* vr, int buf) {
        #pragma unroll
        for (int c = 0; c < 4; ++c) {
            int i  = tid + c * 128;
            int kk = i >> 4;
            int n8 = (i & 15) * 8;
            union { uint4 u; __bf16 h[8]; } vv; vv.u = vr[c];
            #pragma unroll
            for (int j = 0; j < 8; ++j) VT[buf][n8 + j][kk] = vv.h[j];
        }
    };

    // preload Q fragments: {att half} x {k-chunk} -> 4 A fragments
    v16bf qf[4];
    {
        const __bf16* qp = Q + (size_t)(batchBase + qw + l16) * HD;
        #pragma unroll
        for (int f = 0; f < 4; ++f) {
            int base = (f >> 1) * 64 + (f & 1) * 32 + half * 16;
            #pragma unroll
            for (int j = 0; j < 8; ++j) { qf[f][j] = qp[base + j]; qf[f][8 + j] = qp[base + 8 + j]; }
        }
    }

    v8f o1[8], o2[8];
    float m1[8], m2[8], l1[8], l2[8];
    #pragma unroll
    for (int i = 0; i < 8; ++i) {
        #pragma unroll
        for (int j = 0; j < 8; ++j) { o1[i][j] = 0.0f; o2[i][j] = 0.0f; }
        m1[i] = -1e30f; m2[i] = -1e30f; l1[i] = 0.0f; l2[i] = 0.0f;
    }

    const int nT = (qb + BM) / BN;
    uint4 vr[4], vr2[4];
    issueK(0, 0);
    loadV(0, vr);

    for (int t = 0; t < nT; ++t) {
        const int kb  = t * BN;
        const int buf = t & 1;
        storeV(vr, buf);
        if (t + 1 < nT) {
            issueK(t + 1, buf ^ 1);
            loadV(t + 1, vr2);
#if HAVE_ASYNC
            __builtin_amdgcn_s_wait_asynccnt(4);   // tile t complete; t+1 in flight
#endif
        } else {
#if HAVE_ASYNC
            __builtin_amdgcn_s_wait_asynccnt(0);
#endif
        }
        __syncthreads();

        const bool needMask = (kb + BN - 1) > qw;

        #pragma unroll
        for (int att = 0; att < 2; ++att) {
            float* m    = att ? m2 : m1;
            float* lsum = att ? l2 : l1;
            v8f*   o    = att ? o2 : o1;

            // scores S = Q_half . K_half^T  (16 x 32), K-dim = 64 -> 2 WMMAs/tile
            v8f s[2];
            #pragma unroll
            for (int nt = 0; nt < 2; ++nt)
                #pragma unroll
                for (int j = 0; j < 8; ++j) s[nt][j] = 0.0f;
            #pragma unroll
            for (int c = 0; c < 2; ++c) {
                v16bf af = qf[att * 2 + c];
                #pragma unroll
                for (int nt = 0; nt < 2; ++nt) {
                    v16bf bf_;
                    const __bf16* kp = &Ks[buf][nt * 16 + l16][att * 64 + c * 32 + half * 16];
                    #pragma unroll
                    for (int j = 0; j < 16; ++j) bf_[j] = kp[j];
                    s[nt] = __builtin_amdgcn_wmma_f32_16x16x32_bf16(
                        false, af, false, bf_, (short)0, s[nt], false, false);
                }
            }
            if (needMask) {
                #pragma unroll
                for (int nt = 0; nt < 2; ++nt) {
                    int key = kb + nt * 16 + l16;
                    #pragma unroll
                    for (int r = 0; r < 8; ++r)
                        if (key > qw + r + 8 * half) s[nt][r] = -1e30f;
                }
            }

            // online softmax per row (row == accumulator VGPR index), log2 domain
            #pragma unroll
            for (int r = 0; r < 8; ++r) {
                float mx = fmaxf(s[0][r], s[1][r]);
                #pragma unroll
                for (int off = 1; off < 16; off <<= 1) mx = fmaxf(mx, __shfl_xor(mx, off, 32));
                float mn = fmaxf(m[r], mx);
                float sc = __builtin_exp2f(m[r] - mn);
                float p0 = __builtin_exp2f(s[0][r] - mn);
                float p1 = __builtin_exp2f(s[1][r] - mn);
                float rs = p0 + p1;
                #pragma unroll
                for (int off = 1; off < 16; off <<= 1) rs += __shfl_xor(rs, off, 32);
                lsum[r] = lsum[r] * sc + rs;
                m[r] = mn;
                #pragma unroll
                for (int n0 = 0; n0 < 8; ++n0) o[n0][r] *= sc;
                Ps[wave][r + 8 * half][l16]      = f2bf(p0);
                Ps[wave][r + 8 * half][16 + l16] = f2bf(p1);
            }

            // O += P . V   (A-frag from per-wave LDS patch; DS ops are in-order)
            v16bf pa;
            {
                const __bf16* pp = &Ps[wave][l16][half * 16];
                #pragma unroll
                for (int j = 0; j < 8; ++j) { pa[j] = pp[j]; pa[8 + j] = pp[8 + j]; }
            }
            #pragma unroll
            for (int n0 = 0; n0 < 8; ++n0) {
                v16bf vb;
                const __bf16* vp = &VT[buf][n0 * 16 + l16][half * 16];
                #pragma unroll
                for (int j = 0; j < 16; ++j) vb[j] = vp[j];
                o[n0] = __builtin_amdgcn_wmma_f32_16x16x32_bf16(
                    false, pa, false, vb, (short)0, o[n0], false, false);
            }
        }
        __syncthreads();
        #pragma unroll
        for (int c = 0; c < 4; ++c) vr[c] = vr2[c];
    }

    // epilogue: combine halves, RMS-norm over 128 dims, ln_weight, store f32
    float lnw[8];
    #pragma unroll
    for (int n0 = 0; n0 < 8; ++n0) lnw[n0] = ln_w[n0 * 16 + l16];

    #pragma unroll
    for (int r = 0; r < 8; ++r) {
        float i1 = 1.0f / l1[r];
        float i2 = lam / l2[r];
        float ss = 0.0f;
        float val[8];
        #pragma unroll
        for (int n0 = 0; n0 < 8; ++n0) {
            val[n0] = o1[n0][r] * i1 - o2[n0][r] * i2;
            ss += val[n0] * val[n0];
        }
        #pragma unroll
        for (int off = 1; off < 16; off <<= 1) ss += __shfl_xor(ss, off, 32);
        float rinv = (1.0f - LAMBDA_INIT) * rsqrtf(ss * (1.0f / 128.0f) + 1e-5f);
        float* op = out + (size_t)(batchBase + qw + r + 8 * half) * HD;
        #pragma unroll
        for (int n0 = 0; n0 < 8; ++n0)
            op[n0 * 16 + l16] = val[n0] * rinv * lnw[n0];
    }
}

// ---------------------------------------------------------------------------
extern "C" void kernel_launch(void* const* d_in, const int* in_sizes, int n_in,
                              void* d_out, int out_size, void* d_ws, size_t ws_size,
                              hipStream_t stream) {
    const float* x   = (const float*)d_in[0];
    const float* wq  = (const float*)d_in[1];
    const float* wk  = (const float*)d_in[2];
    const float* wv  = (const float*)d_in[3];
    const float* lq1 = (const float*)d_in[4];
    const float* lq2 = (const float*)d_in[5];
    const float* lk1 = (const float*)d_in[6];
    const float* lk2 = (const float*)d_in[7];
    const float* lnw = (const float*)d_in[8];

    const size_t rows = 2u * TLEN;              // 8192
    __bf16* qb = (__bf16*)d_ws;
    __bf16* kb = qb + rows * HD;
    __bf16* vb = kb + rows * HD;
    float*  lam = (float*)(vb + rows * HD);

    proj_kernel<<<dim3(rows / 64, 1, 3), 128, 0, stream>>>(x, wq, wk, wv, qb, kb, vb);
    lambda_kernel<<<1, 64, 0, stream>>>(lq1, lq2, lk1, lk2, lam);
    diffattn_kernel<<<dim3(TLEN / BM, 2), 128, 0, stream>>>(qb, kb, vb, lam, lnw, (float*)d_out);
}
